// SeqAttnLSTM_73564199846362
// MI455X (gfx1250) — compile-verified, matrix operand order checked by reading, not run on hardware
//
#include <hip/hip_runtime.h>
#include <hip/hip_bf16.h>
#include <cstdint>
#include <cstddef>

// Problem dims (from reference)
#define V_ 32000
#define I_ 512
#define H_ 1024
#define B_ 64
#define T_ 64
#define G4 4096          // 4*H
#define NEGBIG (-1.0e9f)

typedef __attribute__((ext_vector_type(16))) __bf16 v16bf;
typedef __attribute__((ext_vector_type(8)))  float  v8f;

union Frag16 { v16bf f; uint4 q[2]; };

// ---------------------------------------------------------------------------
// Register-blocked WMMA GEMM:
//   C(MxN,f32) = A(MxK,bf16,row-major) @ B(NxK,bf16,row-major)^T (+bias[n]) (+Cin)
// Each wave32 computes a 64x16 strip = TM(=4) stacked 16x16 tiles sharing ONE
// B fragment per K-step -> 4x less weight traffic (weights are the roofline
// limiter here: W_o bf16 = 131MB, L2-resident; activations are tiny).
// Requirements: M % 64 == 0, N % 16 == 0, K % 32 == 0 (all call sites comply).
// A fragment per documented 16-bit A layout: lane L -> M = rowbase + (L&15),
// half = L>>4; 16 bf16 = K {half*8..+7} U {16+half*8..+7} => two b128 loads.
// C/D: VGPR i, lane L -> (M = tileM + 8*half + i, N = tileN + (L&15)).
// ---------------------------------------------------------------------------
#define TM 4

__global__ void k_gemm_bf16(const __bf16* __restrict__ A, const __bf16* __restrict__ Bw,
                            const float* __restrict__ bias, const float* __restrict__ Cin,
                            float* __restrict__ Cout,
                            int M, int N, int K, int lda, int ldb, int ldcin, long long ldc)
{
    int wave = threadIdx.x >> 5;
    int lane = threadIdx.x & 31;
    int tiles_n  = N >> 4;
    int strips_m = M >> 6;                       // 64-row strips
    long long strip = (long long)blockIdx.x * (blockDim.x >> 5) + wave;
    long long total = (long long)strips_m * tiles_n;
    if (strip >= total) return;
    int sm = (int)(strip / tiles_n);
    int tn = (int)(strip % tiles_n);
    int l16  = lane & 15;
    int half = lane >> 4;

    int rowbase = sm * 64;                       // first M row of this strip
    const __bf16* ap = A  + (size_t)(rowbase + l16) * lda + half * 8;
    const __bf16* bp = Bw + (size_t)(tn * 16 + l16) * ldb + half * 8;

    int ccol = tn * 16 + l16;

    v8f acc[TM];
#pragma unroll
    for (int mt = 0; mt < TM; ++mt) {
        int crow0 = rowbase + mt * 16 + half * 8;
        if (Cin) {
#pragma unroll
            for (int i = 0; i < 8; ++i) acc[mt][i] = Cin[(size_t)(crow0 + i) * ldcin + ccol];
        } else {
#pragma unroll
            for (int i = 0; i < 8; ++i) acc[mt][i] = 0.0f;
        }
    }
    if (bias) {
        float bv = bias[ccol];
#pragma unroll
        for (int mt = 0; mt < TM; ++mt)
#pragma unroll
            for (int i = 0; i < 8; ++i) acc[mt][i] += bv;
    }

    size_t arow = (size_t)16 * lda;              // 16-row advance within strip

    for (int k = 0; k < K; k += 32) {
        // Pull the streamed weight operand 512 bf16 (16 K-blocks) ahead toward the WGP
        // while current WMMAs execute (lowers to global_prefetch_b8).
        __builtin_prefetch(bp + k + 512, 0, 1);

        Frag16 fb;
        fb.q[0] = *reinterpret_cast<const uint4*>(bp + k);
        fb.q[1] = *reinterpret_cast<const uint4*>(bp + k + 16);
#pragma unroll
        for (int mt = 0; mt < TM; ++mt) {
            Frag16 fa;
            const __bf16* am = ap + (size_t)mt * arow;
            fa.q[0] = *reinterpret_cast<const uint4*>(am + k);
            fa.q[1] = *reinterpret_cast<const uint4*>(am + k + 16);
            acc[mt] = __builtin_amdgcn_wmma_f32_16x16x32_bf16(false, fa.f, false, fb.f,
                                                              (short)0, acc[mt], false, false);
        }
    }

#pragma unroll
    for (int mt = 0; mt < TM; ++mt) {
        int crow0 = rowbase + mt * 16 + half * 8;
#pragma unroll
        for (int i = 0; i < 8; ++i) Cout[(size_t)(crow0 + i) * ldc + ccol] = acc[mt][i];
    }
}

// ---------------------------------------------------------------------------
// Weight/bias prep
// ---------------------------------------------------------------------------
__global__ void k_cvt_bf16(__bf16* dst, const float* src, long long n) {
    long long i = (long long)blockIdx.x * blockDim.x + threadIdx.x;
    if (i < n) dst[i] = (__bf16)src[i];
}

// dst row r = [Wa[r, 0:Ka] | Wb[r, 0:Kb]] converted to bf16
__global__ void k_pack2_bf16(__bf16* dst, const float* Wa, const float* Wb,
                             int Ka, int Kb, long long rows) {
    int K = Ka + Kb;
    long long n = rows * (long long)K;
    long long tid = (long long)blockIdx.x * blockDim.x + threadIdx.x;
    if (tid >= n) return;
    long long r = tid / K;
    int k = (int)(tid % K);
    float v = (k < Ka) ? Wa[r * Ka + k] : Wb[r * Kb + (k - Ka)];
    dst[tid] = (__bf16)v;
}

__global__ void k_bias2(float* dst, const float* a, const float* b, int n) {
    int i = blockIdx.x * blockDim.x + threadIdx.x;
    if (i < n) dst[i] = a[i] + b[i];
}

__global__ void k_zero_f32(float* p, long long n) {
    long long i = (long long)blockIdx.x * blockDim.x + threadIdx.x;
    if (i < n) p[i] = 0.0f;
}
__global__ void k_zero_b16(__bf16* p, long long n) {
    long long i = (long long)blockIdx.x * blockDim.x + threadIdx.x;
    if (i < n) p[i] = (__bf16)0.0f;
}

// zero outputs[:,0,:]
__global__ void k_zero_out0(float* out) {
    long long tid = (long long)blockIdx.x * blockDim.x + threadIdx.x;
    if (tid >= (long long)B_ * V_) return;
    int b = (int)(tid / V_);
    int j = (int)(tid % V_);
    out[(size_t)b * T_ * V_ + j] = 0.0f;
}

// ---------------------------------------------------------------------------
// Encoder input gather: X[(t*B+b)*I + i] = in_emb[src[b, ts]] (time-major, bf16)
// rev!=0 applies the length-reversal index ts = (t<len) ? len-1-t : t
// ---------------------------------------------------------------------------
__global__ void k_gather_x(__bf16* X, const float* emb, const int* src,
                           const int* slen, int rev) {
    long long tid = (long long)blockIdx.x * blockDim.x + threadIdx.x;
    if (tid >= (long long)T_ * B_ * I_) return;
    int t = (int)(tid / (B_ * I_));
    int r = (int)(tid % (B_ * I_));
    int b = r / I_;
    int i = r % I_;
    int ts = t;
    if (rev) { int L = slen[b]; ts = (t < L) ? (L - 1 - t) : t; }
    int tok = src[b * T_ + ts];
    X[tid] = (__bf16)emb[(size_t)tok * I_ + i];
}

// copy time slice t of X into ze[:, 0:512]  (ze row stride 1536)
__global__ void k_copy_x(__bf16* ze, const __bf16* X, int t) {
    int tid = blockIdx.x * blockDim.x + threadIdx.x;
    if (tid >= B_ * I_) return;
    int b = tid / I_;
    int i = tid % I_;
    ze[(size_t)b * 1536 + i] = X[((size_t)t * B_ + b) * I_ + i];
}

// ---------------------------------------------------------------------------
// LSTM pointwise (encoder): masked update, writes bf16 h into ze[:,512:1536]
// ---------------------------------------------------------------------------
__global__ void k_lstm_enc(const float* __restrict__ gates, float* c, float* h32,
                           __bf16* ze, float* outs_t, const int* slen, int t) {
    int tid = blockIdx.x * blockDim.x + threadIdx.x;
    if (tid >= B_ * H_) return;
    int b = tid >> 10;
    int j = tid & 1023;
    const float* g = gates + (size_t)b * G4;
    float gi = g[j], gf = g[H_ + j], gg = g[2 * H_ + j], go = g[3 * H_ + j];
    float si = 1.0f / (1.0f + __expf(-gi));
    float sf = 1.0f / (1.0f + __expf(-gf));
    float so = 1.0f / (1.0f + __expf(-go));
    float cn = sf * c[tid] + si * tanhf(gg);
    float hn = so * tanhf(cn);
    bool m = (t < slen[b]);
    float hc = m ? hn : h32[tid];
    float cc = m ? cn : c[tid];
    h32[tid] = hc;
    c[tid] = cc;
    ze[(size_t)b * 1536 + 512 + j] = (__bf16)hc;
    outs_t[tid] = m ? hn : 0.0f;
}

// LSTM pointwise (decoder): no mask, writes bf16 h into zd[:,1024:2048]
__global__ void k_lstm_dec(const float* __restrict__ gates, float* c, __bf16* zd) {
    int tid = blockIdx.x * blockDim.x + threadIdx.x;
    if (tid >= B_ * H_) return;
    int b = tid >> 10;
    int j = tid & 1023;
    const float* g = gates + (size_t)b * G4;
    float gi = g[j], gf = g[H_ + j], gg = g[2 * H_ + j], go = g[3 * H_ + j];
    float si = 1.0f / (1.0f + __expf(-gi));
    float sf = 1.0f / (1.0f + __expf(-gf));
    float so = 1.0f / (1.0f + __expf(-go));
    float cn = sf * c[tid] + si * tanhf(gg);
    float hn = so * tanhf(cn);
    c[tid] = cn;
    zd[(size_t)b * 2048 + 1024 + j] = (__bf16)hn;
}

// ---------------------------------------------------------------------------
// src_out concat (+ backward un-reversal) -> bf16 (B*T, 2H)
// ---------------------------------------------------------------------------
__global__ void k_concat_enc(__bf16* Z, const float* outf, const float* outbr,
                             const int* slen) {
    long long tid = (long long)blockIdx.x * blockDim.x + threadIdx.x;
    if (tid >= (long long)B_ * T_ * 2048) return;
    int b = (int)(tid / (T_ * 2048));
    int r = (int)(tid % (T_ * 2048));
    int t = r / 2048;
    int j = r % 2048;
    float v;
    if (j < H_) {
        v = outf[((size_t)t * B_ + b) * H_ + j];
    } else {
        int L = slen[b];
        int tp = (t < L) ? (L - 1 - t) : t;
        v = outbr[((size_t)tp * B_ + b) * H_ + (j - H_)];
    }
    Z[tid] = (__bf16)v;
}

// Replicate the reference's stack/reshape: row b<32 takes hT_f rows {2b,2b+1},
// row b>=32 takes hT_b rows {2(b-32), 2(b-32)+1}
__global__ void k_hcat(__bf16* hcat, __bf16* ccat, const float* hf, const float* hb,
                       const float* cf, const float* cb) {
    int tid = blockIdx.x * blockDim.x + threadIdx.x;
    if (tid >= B_ * 2048) return;
    int b = tid >> 11;
    int k = tid & 2047;
    int d = k >> 10;
    int j = k & 1023;
    float hv, cv;
    if (b < 32) { hv = hf[(2 * b + d) * H_ + j];        cv = cf[(2 * b + d) * H_ + j]; }
    else        { hv = hb[(2 * (b - 32) + d) * H_ + j]; cv = cb[(2 * (b - 32) + d) * H_ + j]; }
    hcat[tid] = (__bf16)hv;
    ccat[tid] = (__bf16)cv;
}

__global__ void k_dec_init(__bf16* zd, float* cdec, const float* hid, const float* cel,
                           int* tok, const int* trg) {
    int tid = blockIdx.x * blockDim.x + threadIdx.x;
    if (tid < B_ * H_) {
        int b = tid >> 10;
        int j = tid & 1023;
        zd[(size_t)b * 2048 + 1024 + j] = (__bf16)hid[tid];
        cdec[tid] = cel[tid];
    }
    if (tid < B_) tok[tid] = trg[tid * T_];   // tok0 = trg[:,0]
}

__global__ void k_gather_dec(__bf16* zd, const float* oemb, const int* tok) {
    int tid = blockIdx.x * blockDim.x + threadIdx.x;
    if (tid >= B_ * H_) return;
    int b = tid >> 10;
    int j = tid & 1023;
    zd[(size_t)b * 2048 + j] = (__bf16)oemb[(size_t)tok[b] * H_ + j];
}

// ---------------------------------------------------------------------------
// Attention: one wave32 per (b,t) dot product over H
// ---------------------------------------------------------------------------
__global__ void k_scores(const float* sot, const float* pred, const int* slen, float* sc) {
    int g = blockIdx.x * (blockDim.x >> 5) + (threadIdx.x >> 5);
    if (g >= B_ * T_) return;
    int lane = threadIdx.x & 31;
    int b = g >> 6;
    int t = g & 63;
    const float* sp = sot + (size_t)g * H_;
    const float* pp = pred + (size_t)b * H_;
    float s = 0.0f;
    for (int j = lane; j < H_; j += 32) s += sp[j] * pp[j];
    for (int o = 16; o; o >>= 1) s += __shfl_xor(s, o, 32);
    if (lane == 0) sc[g] = (t < slen[b]) ? s : NEGBIG;
}

// softmax over T=64 with one wave per batch row (2 elems/lane)
__global__ void k_softmax(const float* sc, float* aw) {
    int g = blockIdx.x * (blockDim.x >> 5) + (threadIdx.x >> 5);
    if (g >= B_) return;
    int lane = threadIdx.x & 31;
    float v0 = sc[g * T_ + lane];
    float v1 = sc[g * T_ + 32 + lane];
    float m = fmaxf(v0, v1);
    for (int o = 16; o; o >>= 1) m = fmaxf(m, __shfl_xor(m, o, 32));
    float e0 = __expf(v0 - m);
    float e1 = __expf(v1 - m);
    float s = e0 + e1;
    for (int o = 16; o; o >>= 1) s += __shfl_xor(s, o, 32);
    aw[g * T_ + lane] = e0 / s;
    aw[g * T_ + 32 + lane] = e1 / s;
}

// attn_out + pack z2 = [pred | attn] as bf16 (64 x 2048)
__global__ void k_attn_z2(const float* aw, const float* sot, const float* pred, __bf16* z2) {
    int tid = blockIdx.x * blockDim.x + threadIdx.x;
    if (tid >= B_ * H_) return;
    int b = tid >> 10;
    int j = tid & 1023;
    const float* ab = aw + b * T_;
    const float* sp = sot + ((size_t)b * T_) * H_ + j;
    float s = 0.0f;
    for (int t = 0; t < T_; ++t) s += ab[t] * sp[(size_t)t * H_];
    z2[(size_t)b * 2048 + j] = (__bf16)pred[tid];
    z2[(size_t)b * 2048 + 1024 + j] = (__bf16)s;
}

// ---------------------------------------------------------------------------
// argmax over V per row (LDS tree reduce, first-index tie rule) + TF select
// ---------------------------------------------------------------------------
__global__ void k_argmax_tok(const float* logits, long long ldl, const int* trg,
                             const int* tf_mask, int step, int* tok) {
    __shared__ float sv[256];
    __shared__ int   si[256];
    int b = blockIdx.x;
    const float* row = logits + (size_t)b * ldl;
    float best = -3.4e38f;
    int bi = 0;
    for (int j = threadIdx.x; j < V_; j += 256) {
        float v = row[j];
        if (v > best) { best = v; bi = j; }
    }
    sv[threadIdx.x] = best;
    si[threadIdx.x] = bi;
    __syncthreads();
    for (int s2 = 128; s2 > 0; s2 >>= 1) {
        if ((int)threadIdx.x < s2) {
            float v2 = sv[threadIdx.x + s2];
            int i2 = si[threadIdx.x + s2];
            if (v2 > sv[threadIdx.x] || (v2 == sv[threadIdx.x] && i2 < si[threadIdx.x])) {
                sv[threadIdx.x] = v2;
                si[threadIdx.x] = i2;
            }
        }
        __syncthreads();
    }
    if (threadIdx.x == 0) {
        int top1 = si[0];
        tok[b] = tf_mask[step] ? trg[b * T_ + step + 1] : top1;
    }
}

// ---------------------------------------------------------------------------
// Host orchestration
// ---------------------------------------------------------------------------
#define CDIV(a, b) (((a) + (b) - 1) / (b))

extern "C" void kernel_launch(void* const* d_in, const int* in_sizes, int n_in,
                              void* d_out, int out_size, void* d_ws, size_t ws_size,
                              hipStream_t stream) {
    const int*   src      = (const int*)d_in[0];
    const int*   src_len  = (const int*)d_in[1];
    const int*   trg      = (const int*)d_in[2];
    const int*   tf_mask  = (const int*)d_in[4];
    const float* in_emb   = (const float*)d_in[5];
    const float* out_emb  = (const float*)d_in[6];
    const float* Wih_f    = (const float*)d_in[7];
    const float* Whh_f    = (const float*)d_in[8];
    const float* bih_f    = (const float*)d_in[9];
    const float* bhh_f    = (const float*)d_in[10];
    const float* Wih_b    = (const float*)d_in[11];
    const float* Whh_b    = (const float*)d_in[12];
    const float* bih_b    = (const float*)d_in[13];
    const float* bhh_b    = (const float*)d_in[14];
    const float* W_ht     = (const float*)d_in[15];
    const float* W_ct     = (const float*)d_in[16];
    const float* W_enct   = (const float*)d_in[17];
    const float* dec_Wih  = (const float*)d_in[18];
    const float* dec_Whh  = (const float*)d_in[19];
    const float* dec_bih  = (const float*)d_in[20];
    const float* dec_bhh  = (const float*)d_in[21];
    const float* W_fc     = (const float*)d_in[22];
    const float* W_o      = (const float*)d_in[23];
    float* out = (float*)d_out;

    // ---- workspace partition (all 256B aligned) ----
    char* ws = (char*)d_ws;
    size_t off = 0;
    auto alloc = [&](size_t bytes) -> char* {
        off = (off + 255) & ~(size_t)255;
        char* p = ws + off;
        off += bytes;
        return p;
    };
    __bf16* Wb_enc_f = (__bf16*)alloc((size_t)G4 * 1536 * 2);
    __bf16* Wb_enc_b = (__bf16*)alloc((size_t)G4 * 1536 * 2);
    __bf16* Wb_dec   = (__bf16*)alloc((size_t)G4 * 2048 * 2);
    __bf16* Wb_fc    = (__bf16*)alloc((size_t)H_ * H_ * 2);
    __bf16* Wb_o     = (__bf16*)alloc((size_t)V_ * 2048 * 2);   // 131 MB -> fits L2
    __bf16* Wb_enct  = (__bf16*)alloc((size_t)H_ * 2048 * 2);
    __bf16* Wb_ht    = (__bf16*)alloc((size_t)H_ * 2048 * 2);
    __bf16* Wb_ct    = (__bf16*)alloc((size_t)H_ * 2048 * 2);
    float*  bias_f   = (float*)alloc((size_t)G4 * 4);
    float*  bias_b   = (float*)alloc((size_t)G4 * 4);
    float*  bias_d   = (float*)alloc((size_t)G4 * 4);
    __bf16* X        = (__bf16*)alloc((size_t)T_ * B_ * I_ * 2);
    __bf16* Xr       = (__bf16*)alloc((size_t)T_ * B_ * I_ * 2);
    __bf16* ze_f     = (__bf16*)alloc((size_t)B_ * 1536 * 2);
    __bf16* ze_b     = (__bf16*)alloc((size_t)B_ * 1536 * 2);
    float*  gates    = (float*)alloc((size_t)B_ * G4 * 4);
    float*  c_f      = (float*)alloc((size_t)B_ * H_ * 4);
    float*  c_b      = (float*)alloc((size_t)B_ * H_ * 4);
    float*  h32_f    = (float*)alloc((size_t)B_ * H_ * 4);
    float*  h32_b    = (float*)alloc((size_t)B_ * H_ * 4);
    float*  outs_f   = (float*)alloc((size_t)T_ * B_ * H_ * 4);
    float*  outs_br  = (float*)alloc((size_t)T_ * B_ * H_ * 4);
    __bf16* Zenc     = (__bf16*)alloc((size_t)B_ * T_ * 2048 * 2);
    float*  sot      = (float*)alloc((size_t)B_ * T_ * H_ * 4);  // src_out_t
    __bf16* hcat     = (__bf16*)alloc((size_t)B_ * 2048 * 2);
    __bf16* ccat     = (__bf16*)alloc((size_t)B_ * 2048 * 2);
    float*  hidden32 = (float*)alloc((size_t)B_ * H_ * 4);
    float*  cell32   = (float*)alloc((size_t)B_ * H_ * 4);
    __bf16* zd       = (__bf16*)alloc((size_t)B_ * 2048 * 2);
    float*  c_dec    = (float*)alloc((size_t)B_ * H_ * 4);
    float*  pred     = (float*)alloc((size_t)B_ * H_ * 4);
    float*  sc       = (float*)alloc((size_t)B_ * T_ * 4);
    float*  aw       = (float*)alloc((size_t)B_ * T_ * 4);
    __bf16* z2       = (__bf16*)alloc((size_t)B_ * 2048 * 2);
    int*    tok      = (int*)alloc((size_t)B_ * 4);
    (void)ws_size; (void)in_sizes; (void)n_in; (void)out_size;

    auto gemm = [&](const __bf16* A, const __bf16* Bw, const float* bias,
                    const float* Cin, float* C, int M, int N, int K,
                    int lda, int ldb, int ldcin, long long ldc) {
        int blocks = ((M / 64) * (N / 16) + 7) / 8;   // one 64x16 strip per wave
        k_gemm_bf16<<<blocks, 256, 0, stream>>>(A, Bw, bias, Cin, C, M, N, K,
                                                lda, ldb, ldcin, ldc);
    };

    // ---- one-time weight prep (bf16 path: halves bytes, enables WMMA) ----
    k_pack2_bf16<<<CDIV((long long)G4 * 1536, 256), 256, 0, stream>>>(Wb_enc_f, Wih_f, Whh_f, I_, H_, G4);
    k_pack2_bf16<<<CDIV((long long)G4 * 1536, 256), 256, 0, stream>>>(Wb_enc_b, Wih_b, Whh_b, I_, H_, G4);
    k_pack2_bf16<<<CDIV((long long)G4 * 2048, 256), 256, 0, stream>>>(Wb_dec, dec_Wih, dec_Whh, H_, H_, G4);
    k_cvt_bf16<<<CDIV((long long)H_ * H_, 256), 256, 0, stream>>>(Wb_fc, W_fc, (long long)H_ * H_);
    k_cvt_bf16<<<CDIV((long long)V_ * 2048, 256), 256, 0, stream>>>(Wb_o, W_o, (long long)V_ * 2048);
    k_cvt_bf16<<<CDIV((long long)H_ * 2048, 256), 256, 0, stream>>>(Wb_enct, W_enct, (long long)H_ * 2048);
    k_cvt_bf16<<<CDIV((long long)H_ * 2048, 256), 256, 0, stream>>>(Wb_ht, W_ht, (long long)H_ * 2048);
    k_cvt_bf16<<<CDIV((long long)H_ * 2048, 256), 256, 0, stream>>>(Wb_ct, W_ct, (long long)H_ * 2048);
    k_bias2<<<CDIV(G4, 256), 256, 0, stream>>>(bias_f, bih_f, bhh_f, G4);
    k_bias2<<<CDIV(G4, 256), 256, 0, stream>>>(bias_b, bih_b, bhh_b, G4);
    k_bias2<<<CDIV(G4, 256), 256, 0, stream>>>(bias_d, dec_bih, dec_bhh, G4);

    // ---- embeddings (time-major, bf16), state zeros, out[:,0,:]=0 ----
    k_gather_x<<<CDIV((long long)T_ * B_ * I_, 256), 256, 0, stream>>>(X,  in_emb, src, src_len, 0);
    k_gather_x<<<CDIV((long long)T_ * B_ * I_, 256), 256, 0, stream>>>(Xr, in_emb, src, src_len, 1);
    k_zero_b16<<<CDIV((long long)B_ * 1536, 256), 256, 0, stream>>>(ze_f, (long long)B_ * 1536);
    k_zero_b16<<<CDIV((long long)B_ * 1536, 256), 256, 0, stream>>>(ze_b, (long long)B_ * 1536);
    k_zero_f32<<<CDIV((long long)B_ * H_, 256), 256, 0, stream>>>(c_f,   (long long)B_ * H_);
    k_zero_f32<<<CDIV((long long)B_ * H_, 256), 256, 0, stream>>>(c_b,   (long long)B_ * H_);
    k_zero_f32<<<CDIV((long long)B_ * H_, 256), 256, 0, stream>>>(h32_f, (long long)B_ * H_);
    k_zero_f32<<<CDIV((long long)B_ * H_, 256), 256, 0, stream>>>(h32_b, (long long)B_ * H_);
    k_zero_out0<<<CDIV((long long)B_ * V_, 256), 256, 0, stream>>>(out);

    // ---- encoder forward scan: [x_t | h] @ [Wih|Whh]^T in one WMMA GEMM/step ----
    for (int t = 0; t < T_; ++t) {
        k_copy_x<<<CDIV(B_ * I_, 256), 256, 0, stream>>>(ze_f, X, t);
        gemm(ze_f, Wb_enc_f, bias_f, nullptr, gates, B_, G4, 1536, 1536, 1536, 0, G4);
        k_lstm_enc<<<CDIV(B_ * H_, 256), 256, 0, stream>>>(gates, c_f, h32_f, ze_f,
                                                           outs_f + (size_t)t * B_ * H_, src_len, t);
    }
    // ---- encoder backward scan on reversed input ----
    for (int t = 0; t < T_; ++t) {
        k_copy_x<<<CDIV(B_ * I_, 256), 256, 0, stream>>>(ze_b, Xr, t);
        gemm(ze_b, Wb_enc_b, bias_b, nullptr, gates, B_, G4, 1536, 1536, 1536, 0, G4);
        k_lstm_enc<<<CDIV(B_ * H_, 256), 256, 0, stream>>>(gates, c_b, h32_b, ze_b,
                                                           outs_br + (size_t)t * B_ * H_, src_len, t);
    }

    // ---- src_out_t = concat(out_f, out_b) @ W_enct^T  (4096 x 1024, K=2048) ----
    k_concat_enc<<<CDIV((long long)B_ * T_ * 2048, 256), 256, 0, stream>>>(Zenc, outs_f, outs_br, src_len);
    gemm(Zenc, Wb_enct, nullptr, nullptr, sot, B_ * T_, H_, 2048, 2048, 2048, 0, H_);

    // ---- decoder init: hidden/cell projections ----
    k_hcat<<<CDIV(B_ * 2048, 256), 256, 0, stream>>>(hcat, ccat, h32_f, h32_b, c_f, c_b);
    gemm(hcat, Wb_ht, nullptr, nullptr, hidden32, B_, H_, 2048, 2048, 2048, 0, H_);
    gemm(ccat, Wb_ct, nullptr, nullptr, cell32,   B_, H_, 2048, 2048, 2048, 0, H_);
    k_dec_init<<<CDIV(B_ * H_, 256), 256, 0, stream>>>(zd, c_dec, hidden32, cell32, tok, trg);

    // ---- decoder scan (63 steps); logits GEMM streams bf16 W_o (L2-resident) ----
    for (int s = 0; s < T_ - 1; ++s) {
        k_gather_dec<<<CDIV(B_ * H_, 256), 256, 0, stream>>>(zd, out_emb, tok);
        gemm(zd, Wb_dec, bias_d, nullptr, gates, B_, G4, 2048, 2048, 2048, 0, G4);
        k_lstm_dec<<<CDIV(B_ * H_, 256), 256, 0, stream>>>(gates, c_dec, zd);
        // pred = h_new @ W_fc^T ; A is the h half of zd (lda = 2048)
        gemm(zd + 1024, Wb_fc, nullptr, nullptr, pred, B_, H_, H_, 2048, H_, 0, H_);
        k_scores<<<CDIV(B_ * T_, 8), 256, 0, stream>>>(sot, pred, src_len, sc);
        k_softmax<<<CDIV(B_, 8), 256, 0, stream>>>(sc, aw);
        k_attn_z2<<<CDIV(B_ * H_, 256), 256, 0, stream>>>(aw, sot, pred, z2);
        // logits straight into d_out[:, s+1, :]  (row stride T*V)
        gemm(z2, Wb_o, nullptr, nullptr, out + (size_t)(s + 1) * V_,
             B_, V_, 2048, 2048, 2048, 0, (long long)T_ * V_);
        k_argmax_tok<<<B_, 256, 0, stream>>>(out + (size_t)(s + 1) * V_,
                                             (long long)T_ * V_, trg, tf_mask, s, tok);
    }
}
